// GRURNN_90761248899150
// MI455X (gfx1250) — matrix operand order repaired and checked by reading, not compiled, not run on hardware
//
#include <hip/hip_runtime.h>
#include <hip/hip_bf16.h>
#include <cstdint>
#include <cstddef>

// Problem constants (match reference)
#define T_STEPS 32768
#define IN_DIM  128
#define HID     512
#define GATES   1536   // 3*HID
#define OUT_DIM 128

// Recurrence partitioning
#define NWG_REC      16            // == max workgroups per cluster
#define UNITS_PER_WG 32            // HID / NWG_REC
#define ROWS_PER_WG  96            // 3 * UNITS_PER_WG
#define REC_THREADS  256           // 8 waves
#define ROWS_PER_WAVE 12           // ROWS_PER_WG / 8

typedef __attribute__((ext_vector_type(16))) __bf16 v16bf;
typedef __attribute__((ext_vector_type(8)))  float  v8f;

// ---------------- helpers ----------------

__device__ __forceinline__ unsigned short f2bf(float f) {
  unsigned u = __builtin_bit_cast(unsigned, f);
  u += 0x7FFFu + ((u >> 16) & 1u);            // round-to-nearest-even
  return (unsigned short)(u >> 16);
}
__device__ __forceinline__ float bf2f(unsigned short s) {
  return __builtin_bit_cast(float, (unsigned)s << 16);
}

__global__ void cvt_f32_to_bf16(const float* __restrict__ src,
                                unsigned short* __restrict__ dst, int n) {
  for (int i = blockIdx.x * blockDim.x + threadIdx.x; i < n;
       i += gridDim.x * blockDim.x)
    dst[i] = f2bf(src[i]);
}

// Async VMEM -> LDS b128 copy (ASYNCcnt-tracked; LDS addr = low 32 bits of
// generic pointer since the shared aperture base has zero low bits).
__device__ __forceinline__ void async_copy_b128(unsigned lds_addr, const void* gptr) {
  unsigned long long ga = (unsigned long long)gptr;
  asm volatile("global_load_async_to_lds_b128 %0, %1, off"
               :: "v"(lds_addr), "v"(ga) : "memory");
}

// ---------------- Tensor Data Mover (one-instruction 2D tile DMA) ----------------

#if __has_builtin(__builtin_amdgcn_tensor_load_to_lds)
#define HAVE_TDM 1
typedef unsigned v4u __attribute__((ext_vector_type(4)));
typedef int      v8i __attribute__((ext_vector_type(8)));
typedef int      v4i __attribute__((ext_vector_type(4)));

// 2D tile of 16-bit elements: `rows` x `row_elems`, source row stride
// `stride_elems`, into LDS at byte address `lds_addr`. D# layout per
// CDNA5 ISA (group0: count|lds|global|type,
// group1: data_size|dims|tile|stride). Remaining groups zero (<=2D tensor).
// This toolchain exposes the 6-arg builtin: (g0 v4u, g1 v8i, v4i, v4i, v8i, cpol).
__device__ __forceinline__ void tdm_load_2d(unsigned lds_addr, const void* gptr,
                                            unsigned rows, unsigned row_elems,
                                            unsigned stride_elems) {
  unsigned long long ga = (unsigned long long)gptr;
  v4u g0;
  g0[0] = 1u;                                              // count=1 (user D#)
  g0[1] = lds_addr;                                        // lds_addr [63:32]
  g0[2] = (unsigned)(ga & 0xFFFFFFFFu);                    // global_addr lo
  g0[3] = (unsigned)((ga >> 32) & 0x1FFFFFFu) | (2u << 30);// addr hi | type=2
  v8i g1;
  g1[0] = (int)(1u << 16);                                 // data_size=1 (2B)
  g1[1] = (int)((row_elems & 0xFFFFu) << 16);              // tensor_dim0 lo @48
  g1[2] = (int)(((row_elems >> 16) & 0xFFFFu) |            // tensor_dim0 hi
                ((rows & 0xFFFFu) << 16));                 // tensor_dim1 lo @80
  g1[3] = (int)(((rows >> 16) & 0xFFFFu) |                 // tensor_dim1 hi
                ((row_elems & 0xFFFFu) << 16));            // tile_dim0 @112
  g1[4] = (int)(rows & 0xFFFFu);                           // tile_dim1 (dim2=0)
  g1[5] = (int)stride_elems;                               // tensor_dim0_stride lo
  g1[6] = 0;                                               // stride hi / dim1_stride lo
  g1[7] = 0;                                               // dim1_stride hi
  v4i zz4 = {0, 0, 0, 0};
  v8i zz8 = {0, 0, 0, 0, 0, 0, 0, 0};
  __builtin_amdgcn_tensor_load_to_lds(g0, g1, zz4, zz4, zz8, 0);
}
#else
#define HAVE_TDM 0
#endif

// ---------------- WMMA GEMM: C[M,N] = A[M,K](bf16) @ B[N,K](bf16)^T + bias ----------------
// Block: 256 threads = 8 waves. Each wave: one 16(M) x 128(N) strip -> 8 WMMA
// accumulators. Block tile: 128(M) x 128(N). B K-chunks (128x32 bf16 = 8 KB)
// are double-buffered in LDS via async VMEM->LDS copies: WMMA then depends on
// DScnt (LDS latency) while the next chunk streams in under ASYNCcnt.

union FragBF  { unsigned u[8]; v16bf v; };
union FragBF4 { uint4 q[2]; unsigned u[8]; v16bf v; };

#define BCHUNK_BYTES (128 * 64)   // 128 rows x 32 bf16

__global__ __launch_bounds__(256)
void gemm_bf16_wmma(const unsigned short* __restrict__ A,   // [M,K] row-major bf16
                    const unsigned short* __restrict__ B,   // [N,K] row-major bf16
                    const float* __restrict__ bias,         // [N]
                    float* __restrict__ C,                  // [M,N] f32
                    int M, int N, int K) {
  __shared__ __align__(16) unsigned char bsm[2][BCHUNK_BYTES];

  const int tid  = threadIdx.x;
  const int lane = tid & 31;
  const int wv   = tid >> 5;
  const int m0   = blockIdx.y * 128 + wv * 16;
  const int n0   = blockIdx.x * 128;
  const int half = lane >> 4;     // 0 or 1 (lane group)
  const int l16  = lane & 15;

  v8f c[8];
  #pragma unroll
  for (int j = 0; j < 8; ++j) c[j] = (v8f){};

  const unsigned short* arow = A + (size_t)(m0 + l16) * K;
  const unsigned lds0 = (unsigned)(size_t)&bsm[0][0];
  const unsigned lds1 = (unsigned)(size_t)&bsm[1][0];

  // Issue async copy of B chunk [n0..n0+127] x [kk..kk+31] into buffer `buf`.
  // 512 b128 transfers, 2 per thread.
  auto issue_chunk = [&](int kk, unsigned ldsbase) {
    #pragma unroll
    for (int it = 0; it < 2; ++it) {
      int idx = tid + it * 256;          // 0..511
      int row = idx >> 2;                // 0..127
      int ch  = idx & 3;                 // 16B chunk within 64B row
      async_copy_b128(ldsbase + (unsigned)idx * 16u,
                      B + (size_t)(n0 + row) * K + kk + ch * 8);
    }
  };

  const int nchunks = K / 32;
  issue_chunk(0, lds0);

  for (int ci = 0; ci < nchunks; ++ci) {
    const int kk = ci * 32;
    if (ci + 1 < nchunks) {
      issue_chunk(kk + 32, (ci & 1) ? lds0 : lds1);
      asm volatile("s_wait_asynccnt 0x2" ::: "memory");  // this chunk's 2 done
    } else {
      asm volatile("s_wait_asynccnt 0x0" ::: "memory");
    }
    __syncthreads();   // all waves' copies of chunk ci landed in LDS

    __builtin_prefetch(arow + kk + 64, 0, 3);   // global_prefetch_b8 (near)

    // A fragment: 16x32 bf16, row m = l16, VGPR v holds K pair
    // k0 = half*16 + v*2   (contiguous order, matches B from LDS)
    FragBF a;
    #pragma unroll
    for (int v = 0; v < 8; ++v) {
      int k = kk + half * 16 + v * 2;
      a.u[v] = *(const unsigned*)(arow + k);
    }

    const unsigned char* bb = bsm[ci & 1];
    #pragma unroll
    for (int j = 0; j < 8; ++j) {
      // B fragment from LDS: n-row = j*16+l16, k cols half*16 + 2v,2v+1
      // -> 32 consecutive bytes at row*64 + half*32  (two ds_load_b128)
      const uint4* bp = (const uint4*)(bb + (j * 16 + l16) * 64 + half * 32);
      FragBF4 b;
      b.q[0] = bp[0];
      b.q[1] = bp[1];
      c[j] = __builtin_amdgcn_wmma_f32_16x16x32_bf16(
          false, a.v, false, b.v, (short)0, c[j], false, false);
    }
    __syncthreads();   // done reading bsm[ci&1] before it is refilled
  }

  // C layout: lane 0-15 -> M = v, lane 16-31 -> M = 8+v; N = l16
  #pragma unroll
  for (int j = 0; j < 8; ++j) {
    int n = n0 + j * 16 + l16;
    float bv = bias[n];
    #pragma unroll
    for (int v = 0; v < 8; ++v) {
      int m = m0 + v + 8 * half;
      C[(size_t)m * N + n] = c[j][v] + bv;
    }
  }
}

// ---------------- barriers ----------------

// Atomic count+generation barrier (works under any launch).
__device__ __forceinline__ void grid_sync_atomic(unsigned* bar, unsigned nwg) {
  if (threadIdx.x == 0) {
    unsigned gen  = __hip_atomic_load(bar + 1, __ATOMIC_ACQUIRE, __HIP_MEMORY_SCOPE_AGENT);
    unsigned prev = __hip_atomic_fetch_add(bar, 1u, __ATOMIC_ACQ_REL, __HIP_MEMORY_SCOPE_AGENT);
    if (prev + 1 == nwg) {
      __hip_atomic_store(bar, 0u, __ATOMIC_RELAXED, __HIP_MEMORY_SCOPE_AGENT);
      __hip_atomic_fetch_add(bar + 1, 1u, __ATOMIC_ACQ_REL, __HIP_MEMORY_SCOPE_AGENT);
    } else {
      while (__hip_atomic_load(bar + 1, __ATOMIC_ACQUIRE, __HIP_MEMORY_SCOPE_AGENT) == gen)
        __builtin_amdgcn_s_sleep(2);
    }
  }
}

// Cluster-aware sync: if this dispatch was launched as a 16-WG workgroup
// cluster (IB_STS2.CLUSTER_ID != 0), use the hardware cluster split-barrier
// (s_barrier_signal -3 / s_barrier_wait -3). Otherwise atomic fallback.
__device__ __forceinline__ void rec_sync(unsigned* bar, unsigned nwg, bool in_cluster) {
  __threadfence();
  __syncthreads();
  if (in_cluster) {
    __builtin_amdgcn_s_cluster_barrier();
  } else {
    grid_sync_atomic(bar, nwg);
  }
  __syncthreads();
  __threadfence();
}

__device__ __forceinline__ bool read_in_cluster() {
  unsigned cid;
  asm volatile("s_getreg_b32 %0, hwreg(HW_REG_IB_STS2, 6, 4)" : "=s"(cid));
  return cid != 0;       // 0 => not launched as a cluster
}

// ---------------- persistent GRU recurrence ----------------
// 16 WGs x 256 threads. WG wg owns hidden units [wg*32, wg*32+32): W_hh rows
// {base.., 512+base.., 1024+base..} cached in LDS as bf16 (98 KB of the 320 KB
// WGP LDS), staged by three TDM 2D-tile DMAs (TENSORcnt) or async-LDS copies.
// h double-buffered in global (hbuf[2][512], f32); one cluster/grid barrier
// per timestep.

__global__ __launch_bounds__(REC_THREADS)
void gru_recurrence(const float* __restrict__ ig,          // [T, 1536] f32
                    const unsigned short* __restrict__ Whh,// [1536, 512] bf16
                    const float* __restrict__ bn,          // [512]
                    unsigned short* __restrict__ hout,     // [T, 512] bf16
                    float* __restrict__ hbuf,              // [2, 512] f32 (zeroed)
                    unsigned* __restrict__ bar,            // barrier state (zeroed)
                    int T) {
  extern __shared__ char smem_raw[];
  unsigned short* Ws = (unsigned short*)smem_raw;                     // 96*512 bf16
  float* hsh = (float*)(smem_raw + ROWS_PER_WG * HID * sizeof(unsigned short)); // 512
  float* hg  = hsh + HID;                                             // 96

  const int wg   = blockIdx.x;
  const int base = wg * UNITS_PER_WG;
  const int tid  = threadIdx.x;
  const int lane = tid & 31;
  const int wv   = tid >> 5;
  const bool in_cluster = read_in_cluster();

  // ---- Stage W_hh slice (rows 0..31 = r, 32..63 = z, 64..95 = n) into LDS ----
#if HAVE_TDM
  if (tid < 32) {   // wave 0 issues the DMA (EXEC is ignored by TDM)
    const unsigned lws = (unsigned)(size_t)Ws;
    tdm_load_2d(lws,                Whh + (size_t)(base) * HID,           32, HID, HID);
    tdm_load_2d(lws + 32u * 1024u,  Whh + (size_t)(HID + base) * HID,     32, HID, HID);
    tdm_load_2d(lws + 64u * 1024u,  Whh + (size_t)(2 * HID + base) * HID, 32, HID, HID);
    __builtin_amdgcn_s_wait_tensorcnt(0);
  }
#else
  {
    const unsigned lds_base = (unsigned)(size_t)Ws;
    for (int idx = tid; idx < ROWS_PER_WG * 64; idx += REC_THREADS) {
      int j = idx >> 6;
      int c16 = idx & 63;
      int g = (j < 32) ? (base + j)
            : (j < 64) ? (HID + base + (j - 32))
                       : (2 * HID + base + (j - 64));
      async_copy_b128(lds_base + (unsigned)idx * 16u,
                      Whh + (size_t)g * HID + (size_t)c16 * 8);
    }
    asm volatile("s_wait_asynccnt 0x0" ::: "memory");
  }
#endif
  // h_0 = 0 (hbuf[0] pre-zeroed)
  for (int i = tid; i < HID; i += REC_THREADS) hsh[i] = hbuf[i];
  __syncthreads();

  for (int t = 0; t < T; ++t) {
    // Each lane caches its 16-element h chunk for all rows this step
    float hreg[16];
    #pragma unroll
    for (int q = 0; q < 16; ++q) hreg[q] = hsh[lane * 16 + q];

    // hg[j] = dot(W_row_j, h), wave-parallel rows, lane-parallel K
    for (int rr = 0; rr < ROWS_PER_WAVE; ++rr) {
      int j = wv * ROWS_PER_WAVE + rr;
      const unsigned* wp = (const unsigned*)(Ws + j * HID) + lane * 8;
      float acc = 0.f;
      #pragma unroll
      for (int q = 0; q < 8; ++q) {
        unsigned pk = wp[q];
        acc = fmaf(bf2f((unsigned short)(pk & 0xFFFFu)), hreg[2 * q],     acc);
        acc = fmaf(bf2f((unsigned short)(pk >> 16)),     hreg[2 * q + 1], acc);
      }
      #pragma unroll
      for (int off = 16; off > 0; off >>= 1) acc += __shfl_xor(acc, off, 32);
      if (lane == 0) hg[j] = acc;
    }
    __syncthreads();

    // Gate combine for this WG's 32 units (wave 0)
    if (tid < UNITS_PER_WG) {
      int u = tid;
      const float* igt = ig + (size_t)t * GATES;
      float igr = igt[base + u];
      float igz = igt[HID + base + u];
      float ign = igt[2 * HID + base + u];
      float r = 1.f / (1.f + __expf(-(igr + hg[u])));
      float z = 1.f / (1.f + __expf(-(igz + hg[32 + u])));
      float n = tanhf(ign + r * (hg[64 + u] + bn[base + u]));
      float hp = hsh[base + u];
      float hn = n + z * (hp - n);
      hbuf[((t + 1) & 1) * HID + base + u] = hn;
      hout[(size_t)t * HID + base + u] = f2bf(hn);
    }

    rec_sync(bar, NWG_REC, in_cluster);

    // Pull the full new h (all 16 WGs' slices) into LDS
    const float* hb = hbuf + ((t + 1) & 1) * HID;
    for (int i = tid; i < HID; i += REC_THREADS) hsh[i] = hb[i];
    __syncthreads();
  }
}

// ---------------- host side ----------------

extern "C" void kernel_launch(void* const* d_in, const int* in_sizes, int n_in,
                              void* d_out, int out_size, void* d_ws, size_t ws_size,
                              hipStream_t stream) {
  (void)in_sizes; (void)n_in; (void)out_size; (void)ws_size;

  const float* xs    = (const float*)d_in[0];
  const float* W_ih0 = (const float*)d_in[1];
  const float* W_hh0 = (const float*)d_in[2];
  const float* b0    = (const float*)d_in[3];
  const float* bn0   = (const float*)d_in[4];
  const float* W_ih1 = (const float*)d_in[5];
  const float* W_hh1 = (const float*)d_in[6];
  const float* b1    = (const float*)d_in[7];
  const float* bn1   = (const float*)d_in[8];
  const float* W_lin = (const float*)d_in[9];
  const float* b_lin = (const float*)d_in[10];
  float* out = (float*)d_out;

  // Workspace carve-up (256B aligned)
  char* ws = (char*)d_ws;
  size_t off = 0;
  auto carve = [&](size_t bytes) -> void* {
    void* p = ws + off;
    off = (off + bytes + 255) & ~(size_t)255;
    return p;
  };
  float*          igates = (float*)carve((size_t)T_STEPS * GATES * 4);   // 192 MB (reused both layers)
  unsigned short* h1_bf  = (unsigned short*)carve((size_t)T_STEPS * HID * 2);
  unsigned short* h2_bf  = (unsigned short*)carve((size_t)T_STEPS * HID * 2);
  unsigned short* xs_bf  = (unsigned short*)carve((size_t)T_STEPS * IN_DIM * 2);
  unsigned short* Wih0_b = (unsigned short*)carve((size_t)GATES * IN_DIM * 2);
  unsigned short* Whh0_b = (unsigned short*)carve((size_t)GATES * HID * 2);
  unsigned short* Wih1_b = (unsigned short*)carve((size_t)GATES * HID * 2);
  unsigned short* Whh1_b = (unsigned short*)carve((size_t)GATES * HID * 2);
  unsigned short* Wlin_b = (unsigned short*)carve((size_t)OUT_DIM * HID * 2);
  float*          hbuf   = (float*)carve(2 * HID * 4);
  unsigned*       bar    = (unsigned*)carve(256);

  const int CVT_B = 512, CVT_T = 256;
  cvt_f32_to_bf16<<<CVT_B, CVT_T, 0, stream>>>(xs,    xs_bf,  T_STEPS * IN_DIM);
  cvt_f32_to_bf16<<<CVT_B, CVT_T, 0, stream>>>(W_ih0, Wih0_b, GATES * IN_DIM);
  cvt_f32_to_bf16<<<CVT_B, CVT_T, 0, stream>>>(W_hh0, Whh0_b, GATES * HID);
  cvt_f32_to_bf16<<<CVT_B, CVT_T, 0, stream>>>(W_ih1, Wih1_b, GATES * HID);
  cvt_f32_to_bf16<<<CVT_B, CVT_T, 0, stream>>>(W_hh1, Whh1_b, GATES * HID);
  cvt_f32_to_bf16<<<CVT_B, CVT_T, 0, stream>>>(W_lin, Wlin_b, OUT_DIM * HID);

  const size_t rec_smem = (size_t)ROWS_PER_WG * HID * 2 + HID * 4 + ROWS_PER_WG * 4;

  // ---- Layer 0 ----
  {
    dim3 grid(GATES / 128, T_STEPS / 128);
    gemm_bf16_wmma<<<grid, 256, 0, stream>>>(xs_bf, Wih0_b, b0, igates,
                                             T_STEPS, GATES, IN_DIM);
  }
  (void)hipMemsetAsync(hbuf, 0, 2 * HID * 4, stream);
  (void)hipMemsetAsync(bar, 0, 256, stream);
  gru_recurrence<<<NWG_REC, REC_THREADS, rec_smem, stream>>>(
      igates, Whh0_b, bn0, h1_bf, hbuf, bar, T_STEPS);

  // ---- Layer 1 ----
  {
    dim3 grid(GATES / 128, T_STEPS / 128);
    gemm_bf16_wmma<<<grid, 256, 0, stream>>>(h1_bf, Wih1_b, b1, igates,
                                             T_STEPS, GATES, HID);
  }
  (void)hipMemsetAsync(hbuf, 0, 2 * HID * 4, stream);
  (void)hipMemsetAsync(bar, 0, 256, stream);
  gru_recurrence<<<NWG_REC, REC_THREADS, rec_smem, stream>>>(
      igates, Whh1_b, bn1, h2_bf, hbuf, bar, T_STEPS);

  // ---- Output linear ----
  {
    dim3 grid(OUT_DIM / 128, T_STEPS / 128);
    gemm_bf16_wmma<<<grid, 256, 0, stream>>>(h2_bf, Wlin_b, b_lin, out,
                                             T_STEPS, OUT_DIM, HID);
  }
}
